// TreeNet_78383153152085
// MI455X (gfx1250) — matrix operand order repaired and checked by reading, not compiled, not run on hardware
//
#include <hip/hip_runtime.h>
#include <stdint.h>

// TreeNet scan for MI455X (gfx1250, wave32).
// Strategy: precompute stack-derived gather indices (data-independent), then a
// single persistent kernel: 16 workgroups, each owns a 64x64 tile of the
// [B=256, D=256] state; per step: async global->LDS bf16 child-row gathers +
// on-the-fly X conversion, 48x v_wmma_f32_16x16x32_bf16 per wave (K=768),
// tanh epilogue writing f32 memory (to d_out) + bf16 shadow, then a
// device-wide L2-atomic step barrier.

#define TT   512
#define BB   256
#define DD   256
#define BFN  2
#define NWG  16
#define BM   64
#define BN   64
#define BLOCK 256

typedef __attribute__((ext_vector_type(16))) __bf16 v16bf;
typedef __attribute__((ext_vector_type(8)))  __bf16 v8bf;
typedef __attribute__((ext_vector_type(4)))  __bf16 v4bf;
typedef __attribute__((ext_vector_type(8)))  float  v8f;

__device__ __forceinline__ v16bf cat16(v8bf lo, v8bf hi) {
    return __builtin_shufflevector(lo, hi, 0,1,2,3,4,5,6,7,8,9,10,11,12,13,14,15);
}

// ---------------------------------------------------------------------------
// Kernel 1: simulate the shift-reduce stack per batch lane (depends only on
// arities). Emits child gather indices (TT == "zero row" sentinel) + root idx.
// ---------------------------------------------------------------------------
__global__ void stack_sim(const int* __restrict__ ar,
                          int* __restrict__ cidx,   // [TT][BB][BFN]
                          int* __restrict__ root) { // [BB]
    int b = threadIdx.x;
    int stack[TT + BFN];
    for (int i = 0; i < TT + BFN; ++i) stack[i] = 0;
    int ptr = BFN - 1;
    for (int t = 0; t < TT; ++t) {
        int a = ar[t * BB + b];
        #pragma unroll
        for (int i = 0; i < BFN; ++i) {
            int idx = stack[ptr - i];
            cidx[(((size_t)t * BB) + b) * BFN + i] = (a > i) ? idx : TT; // TT = zero row
        }
        int aa = a < 0 ? -a : a;
        ptr = ptr - aa + 1;
        if (a != -1) stack[ptr] = t;
    }
    root[b] = stack[ptr];
}

// ---------------------------------------------------------------------------
// Kernel 2: persistent scan.
// LDS: Wt [3][BN][DD] bf16 (transposed weight slices, resident) = 96 KB
//      At [3][BM][DD] bf16 (X / child0 / child1 tiles)          = 96 KB
//      bsh[BN] f32 bias slice                                   = 256 B
// ---------------------------------------------------------------------------
__global__ void __launch_bounds__(BLOCK, 1)
treenet_scan(const float* __restrict__ X,     // [TT][BB][DD]
             const float* __restrict__ Win,   // [DD][DD]
             const float* __restrict__ Wch,   // [2][DD][DD]
             const float* __restrict__ bias,  // [DD]
             const int*  __restrict__ cidx,   // [TT][BB][BFN]
             const int*  __restrict__ root,   // [BB]
             float* __restrict__ dout,        // out[BB][DD] ++ memory[TT][BB][DD]
             __bf16* __restrict__ shadow,     // [(TT+1)][BB][DD] bf16
             unsigned int* __restrict__ barctr)
{
    extern __shared__ char smem[];
    __bf16* Wt  = (__bf16*)smem;                          // 3*BN*DD
    __bf16* At  = (__bf16*)(smem + 3 * BN * DD * 2);      // 3*BM*DD
    float*  bsh = (float*)(smem + 6 * BN * DD * 2);       // BN

    const int tid  = threadIdx.x;
    const int wg   = blockIdx.x;
    const int m0   = (wg & 3) * BM;
    const int n0   = (wg >> 2) * BN;
    const int lane = tid & 31;
    const int wv   = tid >> 5;                 // 8 waves
    const int half = lane >> 4;
    const int l15  = lane & 15;

    // One-time: stage bf16-transposed weight slices W^T[n][k] into LDS.
    for (int w = 0; w < 3; ++w) {
        const float* src = (w == 0) ? Win : (Wch + (size_t)(w - 1) * DD * DD);
        for (int i = tid; i < BN * DD; i += BLOCK) {
            int nl = i & (BN - 1);             // fast index -> coalesced reads
            int k  = i >> 6;                   // BN == 64
            Wt[w * BN * DD + nl * DD + k] = (__bf16)src[(size_t)k * DD + n0 + nl];
        }
    }
    if (tid < BN) bsh[tid] = bias[n0 + tid];
    __syncthreads();

    float* memout = dout + BB * DD;            // memory region of d_out
    const uint32_t at_base = (uint32_t)(uintptr_t)At;

    for (int t = 0; t < TT; ++t) {
        // -- stage X tile: f32 -> bf16 (rows m0..m0+63 contiguous) --
        const float4* xs = (const float4*)(X + ((size_t)t * BB + m0) * DD);
        for (int i = tid; i < BM * DD / 4; i += BLOCK) {
            float4 v = xs[i];
            v4bf d; d.x = (__bf16)v.x; d.y = (__bf16)v.y;
                    d.z = (__bf16)v.z; d.w = (__bf16)v.w;
            *(v4bf*)(At + i * 4) = d;
        }
        // -- stage children: 128 bf16 rows, async global->LDS (512 B = 32 lanes x 16 B) --
        for (int rr = wv * 16; rr < wv * 16 + 16; ++rr) {
            int ch = rr >> 6;
            int r  = rr & 63;
            int idx = cidx[((size_t)t * BB + (m0 + r)) * BFN + ch];
            const __bf16* src = shadow + ((size_t)idx * BB + (m0 + r)) * DD + lane * 8;
            uint32_t la = at_base + (uint32_t)((((1 + ch) * BM + r) * DD) * 2 + lane * 16);
            asm volatile("global_load_async_to_lds_b128 %0, %1, off"
                         :: "v"(la), "v"(src) : "memory");
        }
        asm volatile("s_wait_asynccnt 0" ::: "memory");
        __syncthreads();

        // -- GEMM: each wave owns two 16x16 C tiles; K = 3 * 256 = 768 --
        const int wm = (wv & 3) * 16;
        const int wn = (wv >> 2) * 32;
        v8f c0 = {0.f,0.f,0.f,0.f,0.f,0.f,0.f,0.f};
        v8f c1 = c0;
        #pragma unroll
        for (int w = 0; w < 3; ++w) {
            const __bf16* Ab  = At + w * BM * DD + (wm + l15) * DD;
            const __bf16* Bb0 = Wt + w * BN * DD + (wn + l15) * DD;
            const __bf16* Bb1 = Bb0 + 16 * DD;
            #pragma unroll
            for (int k0 = 0; k0 < DD; k0 += 32) {
                // A 16x32 frag: 16 contiguous K at k0 + half*16, row = l15
                v16bf a  = cat16(*(const v8bf*)(Ab + k0 + half * 16),
                                 *(const v8bf*)(Ab + k0 + half * 16 + 8));
                // B 32x16 frags: 16 contiguous K at k0 + half*16, column = l15
                v16bf b0 = cat16(*(const v8bf*)(Bb0 + k0 + half * 16),
                                 *(const v8bf*)(Bb0 + k0 + half * 16 + 8));
                v16bf b1 = cat16(*(const v8bf*)(Bb1 + k0 + half * 16),
                                 *(const v8bf*)(Bb1 + k0 + half * 16 + 8));
                c0 = __builtin_amdgcn_wmma_f32_16x16x32_bf16(false, a, false, b0,
                                                             (short)0, c0, false, false);
                c1 = __builtin_amdgcn_wmma_f32_16x16x32_bf16(false, a, false, b1,
                                                             (short)0, c1, false, false);
            }
        }

        // -- epilogue: bias + tanh; write f32 memory + bf16 shadow --
        #pragma unroll
        for (int r = 0; r < 8; ++r) {
            int m = m0 + wm + (half ? 8 : 0) + r;      // C/D layout per ISA table
            size_t rowoff = ((size_t)t * BB + m) * DD;
            int n  = n0 + wn + l15;
            float v0 = tanhf(c0[r] + bsh[wn + l15]);
            memout[rowoff + n] = v0;
            shadow[rowoff + n] = (__bf16)v0;
            int n2 = n + 16;
            float v1 = tanhf(c1[r] + bsh[wn + 16 + l15]);
            memout[rowoff + n2] = v1;
            shadow[rowoff + n2] = (__bf16)v1;
        }

        // -- device-wide step barrier (monotone L2 atomic counter) --
        __threadfence();
        __syncthreads();
        if (tid == 0) {
            __hip_atomic_fetch_add(barctr, 1u, __ATOMIC_RELEASE, __HIP_MEMORY_SCOPE_AGENT);
            unsigned tgt = (unsigned)(NWG * (t + 1));
            while (__hip_atomic_load(barctr, __ATOMIC_ACQUIRE, __HIP_MEMORY_SCOPE_AGENT) < tgt)
                __builtin_amdgcn_s_sleep(1);
        }
        __syncthreads();
    }

    // -- gather tree roots: out[b,:] = memory[root[b], b, :] --
    for (int i = tid; i < 16 * (DD / 4); i += BLOCK) {
        int b  = wg * 16 + (i >> 6);
        int c4 = i & 63;
        int rt = root[b];
        float4 v = *(const float4*)(memout + ((size_t)rt * BB + b) * DD + c4 * 4);
        *(float4*)(dout + (size_t)b * DD + c4 * 4) = v;
    }
}

// ---------------------------------------------------------------------------
// Host-side launch. Workspace layout:
//   [0,64)        barrier counter
//   [256,1280)    root indices (BB ints)
//   [4096,+1MB)   cidx (TT*BB*BFN ints)
//   [2MB, +64.2MB) bf16 shadow memory [(TT+1)][BB][DD] (slice TT = zeros)
// ---------------------------------------------------------------------------
extern "C" void kernel_launch(void* const* d_in, const int* in_sizes, int n_in,
                              void* d_out, int out_size, void* d_ws, size_t ws_size,
                              hipStream_t stream) {
    (void)in_sizes; (void)n_in; (void)out_size; (void)ws_size;
    const float* X   = (const float*)d_in[0];
    const int*   ar  = (const int*)d_in[1];
    const float* Win = (const float*)d_in[2];
    const float* Wch = (const float*)d_in[3];
    const float* bia = (const float*)d_in[4];
    float* out = (float*)d_out;

    char* ws = (char*)d_ws;
    unsigned* barctr = (unsigned*)(ws);
    int* root = (int*)(ws + 256);
    int* cidx = (int*)(ws + 4096);
    __bf16* shadow = (__bf16*)(ws + (size_t)(2u << 20));

    // Reset barrier counter and the zero-row slice (idx == TT) every call.
    hipMemsetAsync(barctr, 0, 64, stream);
    hipMemsetAsync((char*)shadow + (size_t)TT * BB * DD * 2, 0, (size_t)BB * DD * 2, stream);

    stack_sim<<<1, BB, 0, stream>>>(ar, cidx, root);

    size_t lds_bytes = (size_t)6 * BN * DD * 2 + BN * sizeof(float); // 196864
    treenet_scan<<<NWG, BLOCK, lds_bytes, stream>>>(X, Win, Wch, bia, cidx, root,
                                                    out, shadow, barctr);
}
